// RNN_75531294867647
// MI455X (gfx1250) — matrix operand order repaired and checked by reading, not compiled, not run on hardware
//
#include <hip/hip_runtime.h>
#include <hip/hip_bf16.h>

// ---- types for CDNA5 WMMA ----
typedef __attribute__((ext_vector_type(16))) __bf16 v16bf;
typedef __attribute__((ext_vector_type(8)))  __bf16 v8bf;
typedef __attribute__((ext_vector_type(8)))  float  v8f;

#define NEG_SLOPE 0.01f

// Problem dims
#define DB 128
#define DS 512
#define DI 256
#define DH 1024
#define DO 256

__device__ __forceinline__ v16bf cat16(v8bf lo, v8bf hi) {
  return __builtin_shufflevector(lo, hi, 0, 1, 2, 3, 4, 5, 6, 7,
                                 8, 9, 10, 11, 12, 13, 14, 15);
}

// ---------------- elementwise helpers ----------------
__global__ void cvt_f32_to_bf16(const float* __restrict__ in,
                                __bf16* __restrict__ out) {
  int i = (blockIdx.x * blockDim.x + threadIdx.x) * 4;
  float4 v = *(const float4*)(in + i);
  out[i + 0] = (__bf16)v.x;
  out[i + 1] = (__bf16)v.y;
  out[i + 2] = (__bf16)v.z;
  out[i + 3] = (__bf16)v.w;
}

__global__ void zero_bf16(__bf16* __restrict__ p) {
  int i = blockIdx.x * blockDim.x + threadIdx.x;
  p[i] = (__bf16)0.0f;
}

__global__ void init_sync(unsigned* __restrict__ c, unsigned* __restrict__ g) {
  *c = 0u;
  *g = 0u;
}

// ---------------- generic bf16 WMMA GEMM (parallel phases) ----------------
// D[row,col] = act( sum_k A[row,k] * W[col, kofs+k] + bias[col] + addend[row*add_stride+col] )
// Per-wave tile: 16(M) x 128(N), 8 accumulators; block = 4 waves -> 64 x 128.
// Grid: (N/128, M/64); all dims divide exactly for every call here.
__global__ __launch_bounds__(128)
void gemm_bf16_wmma(const __bf16* __restrict__ A, long lda,
                    const __bf16* __restrict__ W, long ldw, long kofs,
                    const float* __restrict__ bias,
                    const float* __restrict__ addend, long add_stride,
                    float* __restrict__ outF, __bf16* __restrict__ outB, long ldo,
                    int K, int leaky) {
  const int lane  = threadIdx.x & 31;
  const int wave  = threadIdx.x >> 5;
  const int m0    = blockIdx.y * 64 + wave * 16;
  const int n0    = blockIdx.x * 128;
  const int halfA = lane >> 4;
  const int nlane = lane & 15;

  v8f acc[8];
#pragma unroll
  for (int t = 0; t < 8; ++t)
    acc[t] = (v8f){0.f, 0.f, 0.f, 0.f, 0.f, 0.f, 0.f, 0.f};

  const __bf16* Aptr  = A + (long)(m0 + nlane) * lda + halfA * 16;
  const __bf16* Wbase = W + kofs + (long)(n0 + nlane) * ldw + halfA * 16;

  for (int k = 0; k < K; k += 32) {
    v16bf afrag = cat16(*(const v8bf*)(Aptr + k), *(const v8bf*)(Aptr + k + 8));
#pragma unroll
    for (int t = 0; t < 8; ++t) {
      const __bf16* wp = Wbase + (long)t * 16 * ldw + k;
      v16bf bfrag = cat16(*(const v8bf*)(wp), *(const v8bf*)(wp + 8));
      acc[t] = __builtin_amdgcn_wmma_f32_16x16x32_bf16(
          false, afrag, false, bfrag, (short)0, acc[t], false, false);
    }
  }

  const int rowbase = m0 + halfA * 8;
#pragma unroll
  for (int t = 0; t < 8; ++t) {
    const int col = n0 + t * 16 + nlane;
    const float bv = bias ? bias[col] : 0.0f;
#pragma unroll
    for (int r = 0; r < 8; ++r) {
      const long row = rowbase + r;
      float v = acc[t][r] + bv;
      if (addend) v += addend[row * add_stride + col];
      if (leaky) v = (v >= 0.0f) ? v : NEG_SLOPE * v;
      if (outF) outF[row * ldo + col] = v;
      if (outB) outB[row * ldo + col] = (__bf16)v;
    }
  }
}

// ---------------- persistent recurrence kernel ----------------
// 16 blocks x 256 threads (8 waves). Block b owns output columns [b*64, b*64+64).
// Wh2 slice (64 x 1024 bf16 = 128KB) staged in LDS once, reused for all 512 steps.
// Per step: h_{t+1}[m, n] = leaky(G_t[m, n] + sum_k h_t[m,k]*Wh2[n,k] + bh[n])
// Steps separated by a device-wide sense-reversing barrier.
#define STEP_BLOCKS 16
#define STEP_THREADS 256
#define BN 64
#define LDS_STRIDE 1032   // 1024 + 8 pad elements -> 2064B row stride (16B aligned)

__global__ __launch_bounds__(STEP_THREADS)
void rnn_recurrence(const __bf16* __restrict__ Whb,  // [H, 2H]; use cols H..2H-1
                    const float* __restrict__ bh,
                    const float* __restrict__ G,     // [B*S, H] fp32
                    __bf16* __restrict__ hb0,        // ping (holds h0 = 0 on entry)
                    __bf16* __restrict__ hb1,        // pong
                    unsigned* __restrict__ counter,
                    unsigned* __restrict__ gen) {
  __shared__ __bf16 Bsm[BN * LDS_STRIDE];   // 132,096 B < 320KB WGP LDS

  const int tid   = threadIdx.x;
  const int lane  = tid & 31;
  const int wave  = tid >> 5;               // 0..7 -> M rows [wave*16, +16)
  const int n0    = blockIdx.x * BN;
  const int halfA = lane >> 4;
  const int nlane = lane & 15;
  const int m0    = wave * 16;

  // ---- one-time stage of Wh2 slice into LDS ----
  for (int c = tid; c < (BN * DH) / 8; c += STEP_THREADS) {
    const int row = c / (DH / 8);
    const int ce  = (c % (DH / 8)) * 8;
    v8bf v = *(const v8bf*)(Whb + (long)(n0 + row) * (2 * DH) + DH + ce);
    *(v8bf*)(Bsm + row * LDS_STRIDE + ce) = v;
  }
  __syncthreads();

  __bf16* hbuf[2] = {hb0, hb1};
  const __bf16* Bl = Bsm + nlane * LDS_STRIDE + halfA * 16;

  for (int t = 0; t < DS; ++t) {
    const __bf16* A = hbuf[t & 1];
    __bf16* Dst     = hbuf[(t + 1) & 1];

    v8f acc[4];
#pragma unroll
    for (int j = 0; j < 4; ++j)
      acc[j] = (v8f){0.f, 0.f, 0.f, 0.f, 0.f, 0.f, 0.f, 0.f};

    const __bf16* Aptr = A + (long)(m0 + nlane) * DH + halfA * 16;

    for (int k = 0; k < DH; k += 32) {
      v16bf afrag = cat16(*(const v8bf*)(Aptr + k), *(const v8bf*)(Aptr + k + 8));
#pragma unroll
      for (int tt = 0; tt < 4; ++tt) {
        const __bf16* bp = Bl + tt * 16 * LDS_STRIDE + k;
        v16bf bfrag = cat16(*(const v8bf*)(bp), *(const v8bf*)(bp + 8));
        acc[tt] = __builtin_amdgcn_wmma_f32_16x16x32_bf16(
            false, afrag, false, bfrag, (short)0, acc[tt], false, false);
      }
    }

    // epilogue: row = m0 + halfA*8 + r, col = n0 + tt*16 + nlane
    const float* Gt = G + (long)t * DH;
#pragma unroll
    for (int tt = 0; tt < 4; ++tt) {
      const int col = n0 + tt * 16 + nlane;
      const float bv = bh[col];
#pragma unroll
      for (int r = 0; r < 8; ++r) {
        const long row = m0 + halfA * 8 + r;
        float v = acc[tt][r] + bv + Gt[row * (long)(DS * DH) + col];
        v = (v >= 0.0f) ? v : NEG_SLOPE * v;
        Dst[row * DH + col] = (__bf16)v;
      }
    }

    // ---- device-wide barrier (sense-reversing, monotonic generation) ----
    __threadfence();          // release this block's h writes to device scope
    __syncthreads();
    if (tid == 0) {
      const unsigned g = *((volatile unsigned*)gen);
      const unsigned arrived = atomicAdd(counter, 1u) + 1u;
      if (arrived == STEP_BLOCKS) {
        __threadfence();
        atomicExch(counter, 0u);  // reset before releasing waiters
        __threadfence();
        atomicAdd(gen, 1u);
      } else {
        while (*((volatile unsigned*)gen) == g) __builtin_amdgcn_s_sleep(8);
      }
    }
    __syncthreads();
    __threadfence();          // acquire: invalidate stale h lines before next step
  }
}

// ---------------- launcher ----------------
extern "C" void kernel_launch(void* const* d_in, const int* in_sizes, int n_in,
                              void* d_out, int out_size, void* d_ws, size_t ws_size,
                              hipStream_t stream) {
  const float* x  = (const float*)d_in[0];  // [B,S,I]
  const float* Wi = (const float*)d_in[1];  // [H,I]
  const float* bi = (const float*)d_in[2];  // [H]
  const float* Wh = (const float*)d_in[3];  // [H,2H]
  const float* bh = (const float*)d_in[4];  // [H]
  const float* Wo = (const float*)d_in[5];  // [O,H]
  const float* bo = (const float*)d_in[6];  // [O]
  float* out = (float*)d_out;               // [B,O]

  const long BS = (long)DB * DS;

  // workspace carve-up (all offsets multiples of 256B)
  char* ws = (char*)d_ws;
  size_t off = 0;
  __bf16* xb     = (__bf16*)(ws + off); off += (size_t)BS * DI * 2;       // 32 MB
  __bf16* Wib    = (__bf16*)(ws + off); off += (size_t)DH * DI * 2;       // 0.5 MB
  __bf16* Whb    = (__bf16*)(ws + off); off += (size_t)DH * 2 * DH * 2;   // 4 MB
  __bf16* Wob    = (__bf16*)(ws + off); off += (size_t)DO * DH * 2;       // 0.5 MB
  __bf16* xprojb = (__bf16*)(ws + off); off += (size_t)BS * DH * 2;       // 128 MB
  float*  G      = (float*) (ws + off); off += (size_t)BS * DH * 4;       // 256 MB
  __bf16* hb0    = (__bf16*)(ws + off); off += (size_t)DB * DH * 2;       // 256 KB
  __bf16* hb1    = (__bf16*)(ws + off); off += (size_t)DB * DH * 2;       // 256 KB
  unsigned* sync = (unsigned*)(ws + off); off += 256;                     // counter, gen

  // 1) fp32 -> bf16 conversions (x4 vectorized; all counts % 1024 == 0)
  {
    long n;
    n = BS * DI;           cvt_f32_to_bf16<<<dim3(n / 1024), dim3(256), 0, stream>>>(x,  xb);
    n = (long)DH * DI;     cvt_f32_to_bf16<<<dim3(n / 1024), dim3(256), 0, stream>>>(Wi, Wib);
    n = (long)DH * 2 * DH; cvt_f32_to_bf16<<<dim3(n / 1024), dim3(256), 0, stream>>>(Wh, Whb);
    n = (long)DO * DH;     cvt_f32_to_bf16<<<dim3(n / 1024), dim3(256), 0, stream>>>(Wo, Wob);
  }

  // 2) xproj = leaky(x @ Wi^T + bi)   [BS, H] (bf16)
  gemm_bf16_wmma<<<dim3(DH / 128, BS / 64), dim3(128), 0, stream>>>(
      xb, DI, Wib, DI, 0, bi, nullptr, 0, nullptr, xprojb, DH, DI, 1);

  // 3) G = xproj @ Wh1^T              [BS, H] (fp32, no act/bias)
  gemm_bf16_wmma<<<dim3(DH / 128, BS / 64), dim3(128), 0, stream>>>(
      xprojb, DH, Whb, 2 * DH, 0, nullptr, nullptr, 0, G, nullptr, DH, DH, 0);

  // 4) h0 = 0 ; barrier state = 0
  zero_bf16<<<dim3((DB * DH) / 256), dim3(256), 0, stream>>>(hb0);
  init_sync<<<dim3(1), dim3(1), 0, stream>>>(sync, sync + 1);

  // 5) persistent recurrence: 512 steps, weights pinned in LDS, grid barrier/step
  rnn_recurrence<<<dim3(STEP_BLOCKS), dim3(STEP_THREADS), 0, stream>>>(
      Whb, bh, G, hb0, hb1, sync, sync + 1);

  // 6) out = h_final @ Wo^T + bo      [B, O] fp32  (DS even -> final h in hb0)
  gemm_bf16_wmma<<<dim3(DO / 128, DB / 64), dim3(128), 0, stream>>>(
      hb0, DH, Wob, DH, 0, bo, nullptr, 0, out, nullptr, DO, DH, 0);
}
